// CrossFrameAttentionCal_45758581571627
// MI455X (gfx1250) — matrix operand order repaired and checked by reading, not compiled, not run on hardware
//
#include <hip/hip_runtime.h>
#include <hip/hip_bf16.h>

#define DIMC   1024
#define NHEADS 16
#define BATCH  2
#define SEQ    2048
#define BS     (BATCH * SEQ)   // 4096 rows total

typedef __attribute__((ext_vector_type(16))) __bf16 v16bf;
typedef __attribute__((ext_vector_type(8)))  __bf16 v8bf;
typedef __attribute__((ext_vector_type(8)))  float  v8f;
typedef __attribute__((ext_vector_type(4)))  float  v4f;
typedef __attribute__((ext_vector_type(4)))  unsigned int v4u;
typedef __attribute__((ext_vector_type(8)))  int    v8i;
typedef __attribute__((ext_vector_type(4)))  int    v4i;

#if __has_builtin(__builtin_amdgcn_tensor_load_to_lds) && \
    __has_builtin(__builtin_amdgcn_s_wait_tensorcnt)
#define HAVE_TDM 1
#endif

// ---- ISA 7.12.2 fragment index maps (wave32, 16-bit data) -------------------
// A (16x32 bf16): lane L holds row M = L&15; element e -> K:
//   lanes 0-15 : e0..7 -> K=e,    e8..15 -> K=e+8   (16..23)
//   lanes 16-31: e0..7 -> K=e+8,  e8..15 -> K=e+16  (24..31)
// => two contiguous 8-element runs at K = 8*half and K = 16 + 8*half.
// B (32x16 bf16): lane L holds col N = L&15; A and B here: e -> K = e + 16*half (ISA's B map is A's)
// => one contiguous 16-element run at K = 16*half.
// C/D (16x16 f32): VGPR r, lane L -> row M = r + 8*half, col N = L&15

__device__ __forceinline__ v16bf loadAfrag_bf16(const __bf16* base, int half) {
    v8bf lo = *(const v8bf*)(base + 16 * half);
    v8bf hi = *(const v8bf*)(base + 16 * half + 8);
    v16bf a;
#pragma unroll
    for (int e = 0; e < 8; ++e) { a[e] = lo[e]; a[e + 8] = hi[e]; }
    return a;
}

__device__ __forceinline__ v16bf loadAfrag_f32(const float* base, int half) {
    v4f lo0 = *(const v4f*)(base + 16 * half);
    v4f lo1 = *(const v4f*)(base + 16 * half + 4);
    v4f hi0 = *(const v4f*)(base + 16 * half + 8);
    v4f hi1 = *(const v4f*)(base + 16 * half + 12);
    v16bf a;
#pragma unroll
    for (int e = 0; e < 4; ++e) {
        a[e]      = (__bf16)lo0[e];
        a[e + 4]  = (__bf16)lo1[e];
        a[e + 8]  = (__bf16)hi0[e];
        a[e + 12] = (__bf16)hi1[e];
    }
    return a;
}

#define WMMA_BF16(A, B, C) \
    __builtin_amdgcn_wmma_f32_16x16x32_bf16(false, (A), false, (B), (short)0, (C), false, false)

// ---- Projection GEMM: out[BS,N] = bf16( X[BS,1024] @ W[1024,N] + bias ) -----
// 2x2 register blocking: each wave computes a 32x32 tile (4 WMMAs / K-step).
__global__ void __launch_bounds__(128)
proj_gemm_bf16(const float* __restrict__ X, const float* __restrict__ W,
               const float* __restrict__ bias, __bf16* __restrict__ out, int N) {
    const int lane = threadIdx.x & 31;
    const int wave = threadIdx.x >> 5;
    const int half = lane >> 4;
    const int l16  = lane & 15;
    const int mt2  = blockIdx.x;              // 32-row tile
    const int nt2  = blockIdx.y * 4 + wave;   // 32-col tile
    const long row0 = (long)mt2 * 32 + l16, row1 = row0 + 16;
    const long col0 = (long)nt2 * 32 + l16, col1 = col0 + 16;

    v8f acc00 = {}, acc01 = {}, acc10 = {}, acc11 = {};
    for (int kb = 0; kb < DIMC; kb += 32) {
        v16bf a0 = loadAfrag_f32(X + row0 * DIMC + kb, half);
        v16bf a1 = loadAfrag_f32(X + row1 * DIMC + kb, half);
        v16bf b0, b1;
#pragma unroll
        for (int e = 0; e < 16; ++e) {         // lane-coalesced column reads
            const float* wr = W + (long)(kb + e + 16 * half) * N;
            b0[e] = (__bf16)wr[col0];
            b1[e] = (__bf16)wr[col1];
        }
        acc00 = WMMA_BF16(a0, b0, acc00);
        acc01 = WMMA_BF16(a0, b1, acc01);
        acc10 = WMMA_BF16(a1, b0, acc10);
        acc11 = WMMA_BF16(a1, b1, acc11);
    }
    const float bv0 = bias[col0], bv1 = bias[col1];
#pragma unroll
    for (int r = 0; r < 8; ++r) {
        const long orow0 = (long)mt2 * 32 + r + 8 * half;
        const long orow1 = orow0 + 16;
        out[orow0 * N + col0] = (__bf16)(acc00[r] + bv0);
        out[orow0 * N + col1] = (__bf16)(acc01[r] + bv1);
        out[orow1 * N + col0] = (__bf16)(acc10[r] + bv0);
        out[orow1 * N + col1] = (__bf16)(acc11[r] + bv1);
    }
}

// ---- mo = cr @ Wmo + bmo (rank-2, pure bandwidth, 8 elems/thread) -----------
__global__ void mo_proj(const float* __restrict__ cr, const float* __restrict__ Wmo,
                        const float* __restrict__ bmo, __bf16* __restrict__ mob) {
    const long idx8 = (long)blockIdx.x * blockDim.x + threadIdx.x; // over BS*DIMC/8
    const long s = idx8 >> 7;                  // DIMC/8 = 128 chunks per row
    const int  j = (int)(idx8 & 127) * 8;
    const float c0 = cr[s * 2], c1 = cr[s * 2 + 1];
    v8bf o;
#pragma unroll
    for (int i = 0; i < 8; ++i)
        o[i] = (__bf16)(c0 * Wmo[j + i] + c1 * Wmo[DIMC + j + i] + bmo[j + i]);
    *(v8bf*)(mob + s * DIMC + j) = o;
}

// ---- Fused flash attention with dual value streams (V and MO) ---------------
// grid.x = B * H * (SEQ/64); block = 128 (4 waves, 16 q-rows each).
// The 32-key K tile is DMA'd to LDS by the Tensor Data Mover (wave 0 issues the
// 2-D descriptor, waits TENSORcnt, barrier releases the block); V/MO are staged
// transposed by the vector pipes so PV B-fragments are contiguous LDS reads.
__global__ void __launch_bounds__(128)
flash_attn_dual(const __bf16* __restrict__ qb,   // [BS, 1024]
                const __bf16* __restrict__ kvb,  // [BS, 2048] (K | V)
                const __bf16* __restrict__ mob,  // [BS, 1024]
                float* __restrict__ out_h, float* __restrict__ out_m) {
    const int lane = threadIdx.x & 31;
    const int wave = threadIdx.x >> 5;
    const int half = lane >> 4;
    const int l16  = lane & 15;

    const int bid  = blockIdx.x;
    const int qblk = bid & 31;          // SEQ/64 = 32
    const int h    = (bid >> 5) & 15;
    const int b    = bid >> 9;
    const long srow0 = (long)b * SEQ;
    const int  qr    = qblk * 64 + wave * 16;

    __shared__ __bf16 Kt[32][64];       // [key][d]    4 KB (TDM destination)
    __shared__ __bf16 VtT[64][32];      // [d][key]    4 KB (transposed)
    __shared__ __bf16 MtT[64][32];      // [d][key]    4 KB (transposed)
    __shared__ __bf16 ldsP[4][16][32];  // per-wave P  4 KB

    // Q A-fragments: d = 0..31 and 32..63 (two b128 loads each)
    const __bf16* qrow = qb + (srow0 + qr + l16) * DIMC + h * 64;
    const v16bf aq0 = loadAfrag_bf16(qrow, half);
    const v16bf aq1 = loadAfrag_bf16(qrow + 32, half);

    float m_i[8], l_i[8];
    v8f acc_h[4], acc_m[4];
#pragma unroll
    for (int r = 0; r < 8; ++r) { m_i[r] = -3.0e38f; l_i[r] = 0.0f; }
#pragma unroll
    for (int d0 = 0; d0 < 4; ++d0) { acc_h[d0] = (v8f){}; acc_m[d0] = (v8f){}; }

    const int trow = threadIdx.x >> 2;  // 0..31: key row this thread stages
    const int tseg = threadIdx.x & 3;   // 0..3 : 16-wide d segment

    for (int kt = 0; kt < SEQ / 32; ++kt) {
        const int kbase = kt * 32;
        const long krow = srow0 + kbase + trow;
        const __bf16* src = kvb + krow * (2 * DIMC) + h * 64;

        // ---- K tile via Tensor Data Mover (2-D descriptor, TENSORcnt) -------
#ifdef HAVE_TDM
        if (wave == 0) {
            const unsigned long long ga =
                (unsigned long long)(const void*)(kvb + (srow0 + kbase) * (2 * DIMC) + h * 64);
            const unsigned int lds_off =
                (unsigned int)(unsigned long long)(const void*)&Kt[0][0];
            v4u g0;
            g0[0] = 1u;                                   // count=1 (valid, user)
            g0[1] = lds_off;                              // lds_addr [63:32]
            g0[2] = (unsigned int)(ga & 0xFFFFFFFFu);     // global_addr lo
            g0[3] = (unsigned int)((ga >> 32) & 0x01FFFFFFu) | (2u << 30); // ga hi | type=2
            v8i g1;
            g1[0] = 0x00010000;                           // data_size=2B, no multicast
            g1[1] = (int)(64u << 16);                     // tensor_dim0 = 64 (lo16)
            g1[2] = (int)(32u << 16);                     // dim0 hi16=0, tensor_dim1 = 32
            g1[3] = (int)(64u << 16);                     // dim1 hi16=0, tile_dim0 = 64
            g1[4] = 32;                                   // tile_dim1 = 32, tile_dim2 = 0
            g1[5] = 2 * DIMC;                             // tensor_dim0_stride = 2048
            g1[6] = 0;                                    // stride0 hi, stride1 lo
            g1[7] = 0;                                    // stride1 hi
            const v4i gz4 = {};
            const v8i gz8 = {};
            __builtin_amdgcn_tensor_load_to_lds(g0, g1, gz4, gz4, gz8, 0);
            __builtin_amdgcn_s_wait_tensorcnt(0);
        }
#else
        *(v16bf*)&Kt[trow][tseg * 16] = *(const v16bf*)(src + tseg * 16);
#endif

        // ---- V/MO staged transposed by all 128 threads ----------------------
        {
            const v16bf vv = *(const v16bf*)(src + DIMC + tseg * 16);
            const v16bf mm = *(const v16bf*)(mob + krow * DIMC + h * 64 + tseg * 16);
#pragma unroll
            for (int i = 0; i < 16; ++i) {
                VtT[tseg * 16 + i][trow] = vv[i];
                MtT[tseg * 16 + i][trow] = mm[i];
            }
        }
        // prefetch next key tile into L2 while this one is consumed
        if (kt + 1 < SEQ / 32) {
            const __bf16* nsrc = src + 32 * (2 * DIMC);
            __builtin_prefetch(nsrc, 0, 3);
            __builtin_prefetch(nsrc + DIMC, 0, 3);
            __builtin_prefetch(mob + (krow + 32) * DIMC + h * 64 + tseg * 16, 0, 3);
        }
        __syncthreads();

        // ---- scores: S0 = keys 0..15, S1 = keys 16..31 ----------------------
        v8f s0 = {}, s1 = {};
        s0 = WMMA_BF16(aq0, *(const v16bf*)&Kt[l16][16 * half],           s0);
        s0 = WMMA_BF16(aq1, *(const v16bf*)&Kt[l16][32 + 16 * half],      s0);
        s1 = WMMA_BF16(aq0, *(const v16bf*)&Kt[16 + l16][16 * half],      s1);
        s1 = WMMA_BF16(aq1, *(const v16bf*)&Kt[16 + l16][32 + 16 * half], s1);

        // ---- online softmax (row stats reduced across the 16-lane half) -----
#pragma unroll
        for (int r = 0; r < 8; ++r) {
            const float v0 = s0[r] * 0.125f;   // 1/sqrt(64)
            const float v1 = s1[r] * 0.125f;
            float mx = fmaxf(v0, v1);
#pragma unroll
            for (int off = 1; off < 16; off <<= 1) mx = fmaxf(mx, __shfl_xor(mx, off));
            const float mn = fmaxf(m_i[r], mx);
            const float sc = __expf(m_i[r] - mn);
            m_i[r] = mn;
            const float p0 = __expf(v0 - mn);
            const float p1 = __expf(v1 - mn);
            float rs = p0 + p1;
#pragma unroll
            for (int off = 1; off < 16; off <<= 1) rs += __shfl_xor(rs, off);
            l_i[r] = l_i[r] * sc + rs;
#pragma unroll
            for (int d0 = 0; d0 < 4; ++d0) { acc_h[d0][r] *= sc; acc_m[d0][r] *= sc; }
            ldsP[wave][r + 8 * half][l16]      = (__bf16)p0;
            ldsP[wave][r + 8 * half][16 + l16] = (__bf16)p1;
        }
        asm volatile("s_wait_dscnt 0" ::: "memory");   // wave-local LDS RAW

        // P as A-fragment (contraction over the 32 keys)
        const v16bf ap = loadAfrag_bf16(&ldsP[wave][l16][0], half);

        // ---- O_h += P @ V , O_m += P @ MO (contiguous LDS B-frags) ----------
#pragma unroll
        for (int d0 = 0; d0 < 4; ++d0) {
            const v16bf bv = *(const v16bf*)&VtT[d0 * 16 + l16][16 * half];
            const v16bf bm = *(const v16bf*)&MtT[d0 * 16 + l16][16 * half];
            acc_h[d0] = WMMA_BF16(ap, bv, acc_h[d0]);
            acc_m[d0] = WMMA_BF16(ap, bm, acc_m[d0]);
        }
        __syncthreads();   // protect Kt/VtT/MtT before next stage
    }

    // ---- epilogue: divide by l, write f32 outputs ---------------------------
#pragma unroll
    for (int r = 0; r < 8; ++r) {
        const float inv = 1.0f / l_i[r];
        const int   q   = qr + r + 8 * half;
        const long  o   = (srow0 + q) * (long)DIMC + h * 64;
#pragma unroll
        for (int d0 = 0; d0 < 4; ++d0) {
            out_h[o + d0 * 16 + l16] = acc_h[d0][r] * inv;
            out_m[o + d0 * 16 + l16] = acc_m[d0][r] * inv;
        }
    }
}

extern "C" void kernel_launch(void* const* d_in, const int* in_sizes, int n_in,
                              void* d_out, int out_size, void* d_ws, size_t ws_size,
                              hipStream_t stream) {
    (void)in_sizes; (void)n_in; (void)out_size; (void)ws_size;
    const float* i1  = (const float*)d_in[0];
    const float* i2  = (const float*)d_in[1];
    const float* cr  = (const float*)d_in[2];
    const float* Wq  = (const float*)d_in[3];
    const float* bq  = (const float*)d_in[4];
    const float* Wkv = (const float*)d_in[5];
    const float* bkv = (const float*)d_in[6];
    const float* Wmo = (const float*)d_in[7];
    const float* bmo = (const float*)d_in[8];
    float* out = (float*)d_out;

    char* ws = (char*)d_ws;
    __bf16* qbuf  = (__bf16*)ws;                           //  8 MB: [BS,1024]
    __bf16* kvbuf = (__bf16*)(ws + ((size_t)8  << 20));    // 16 MB: [BS,2048]
    __bf16* mobuf = (__bf16*)(ws + ((size_t)24 << 20));    //  8 MB: [BS,1024]

    dim3 blk(128);
    // Q projection: 128 (32-row tiles) x 8 (32-col tile groups of 4)
    proj_gemm_bf16<<<dim3(128, 8), blk, 0, stream>>>(i1, Wq, bq, qbuf, DIMC);
    // KV projection: 128 x 16
    proj_gemm_bf16<<<dim3(128, 16), blk, 0, stream>>>(i2, Wkv, bkv, kvbuf, 2 * DIMC);
    // MO projection (rank-2), 8 bf16 per thread
    mo_proj<<<(BS * DIMC / 8) / 256, 256, 0, stream>>>(cr, Wmo, bmo, mobuf);
    // Fused dual-stream flash attention: B*H*(SEQ/64) = 1024 blocks
    flash_attn_dual<<<BATCH * NHEADS * (SEQ / 64), blk, 0, stream>>>(
        qbuf, kvbuf, mobuf, out, out + (size_t)BS * DIMC);
}
